// Multi_head_attention_55018531062080
// MI455X (gfx1250) — matrix operand order repaired and checked by reading, not compiled, not run on hardware
//
#include <hip/hip_runtime.h>
#include <stdint.h>

// ---------------------------------------------------------------------------
// Transformer-XL relative multi-head attention for MI455X (gfx1250, wave32).
// bf16 WMMA everywhere; GEMM tiles staged via async Global->LDS DMA
// (GLOBAL_LOAD_ASYNC_TO_LDS_B128 + s_wait_asynccnt). Workspace ~320 MB.
// ---------------------------------------------------------------------------

typedef __attribute__((ext_vector_type(16))) __bf16 v16bf;
typedef __attribute__((ext_vector_type(8)))  float  v8f;
typedef __attribute__((ext_vector_type(4)))  unsigned int u32x4;

union Frag {
  v16bf  v;
  u32x4  q[2];
  __bf16 e[16];
};

__device__ __forceinline__ v8f wmma_bf16(v16bf a, v16bf b, v8f c) {
  return __builtin_amdgcn_wmma_f32_16x16x32_bf16(false, a, false, b, (short)0, c,
                                                 false, false);
}

// Async 16B Global -> LDS copy (per-lane), tracked by ASYNCcnt.
// Low 32 bits of a flat pointer to LDS == wave-relative LDS byte offset.
__device__ __forceinline__ void async_ld_b128(void* lds, const void* g) {
  uint32_t l = (uint32_t)(uintptr_t)lds;
  asm volatile("global_load_async_to_lds_b128 %0, %1, off"
               :: "v"(l), "v"(g) : "memory");
}
template <int N>
__device__ __forceinline__ void wait_asynccnt() {
  asm volatile("s_wait_asynccnt %0" :: "n"(N) : "memory");
}

// ---------------------------------------------------------------------------
// Elementwise f32 -> bf16
// ---------------------------------------------------------------------------
__global__ __launch_bounds__(256)
void convert_bf16(const float* __restrict__ src, __bf16* __restrict__ dst) {
  size_t i = (size_t)blockIdx.x * 256 + threadIdx.x;
  dst[i] = (__bf16)src[i];
}

// Fused concat(mem, x) -> bf16 x_tilde  (B=2, MEM=SEQ=1024, D=1024)
__global__ __launch_bounds__(256)
void build_xtb(const float* __restrict__ x, const float* __restrict__ memp,
               __bf16* __restrict__ xtb) {
  size_t idx = (size_t)blockIdx.x * 256 + threadIdx.x;  // 2*2048*1024
  int    col = (int)(idx & 1023);
  size_t row = idx >> 10;
  int    t   = (int)(row & 2047);
  size_t b   = row >> 11;
  float v = (t < 1024) ? memp[(b * 1024 + t) * 1024 + col]
                       : x[(b * 1024 + (t - 1024)) * 1024 + col];
  xtb[idx] = (__bf16)v;
}

// Weight convert+transpose: W[k][n] f32 -> Wt[n][k] bf16 (1024x1024), LDS tiled
__global__ __launch_bounds__(256)
void wtrans_bf16(const float* __restrict__ W, __bf16* __restrict__ Wt) {
  __shared__ __bf16 tile[32][33];
  int tx = threadIdx.x & 31;
  int ty = threadIdx.x >> 5;        // 0..7
  int k0 = blockIdx.x * 32;
  int n0 = blockIdx.y * 32;
#pragma unroll
  for (int i = 0; i < 4; ++i) {
    int kk = ty + i * 8;
    tile[kk][tx] = (__bf16)W[(size_t)(k0 + kk) * 1024 + n0 + tx];
  }
  __syncthreads();
#pragma unroll
  for (int i = 0; i < 4; ++i) {
    int nn = ty + i * 8;
    Wt[(size_t)(n0 + nn) * 1024 + k0 + tx] = tile[tx][nn];
  }
}

// ---------------------------------------------------------------------------
// GEMM: Y[M x 1024] = A[M x 1024](bf16) @ Wt[n][k](bf16, pre-transposed) + bias
// Block 128x128, BK=64, 8 waves x (32x64). Tiles staged with async DMA into
// double-buffered LDS; wave only waits (s_wait_asynccnt) and barriers.
// ---------------------------------------------------------------------------
template <typename OutT>
__global__ __launch_bounds__(256)
void gemm1024_bf16(const __bf16* __restrict__ A, const __bf16* __restrict__ Bt,
                   const float* __restrict__ bias, OutT* __restrict__ Y) {
  __shared__ __align__(16) __bf16 As[2][128 * 64];
  __shared__ __align__(16) __bf16 Bs[2][128 * 64];

  const int tid  = threadIdx.x;
  const int lane = tid & 31;
  const int ln   = lane & 15;
  const int g    = lane >> 4;
  const int w    = tid >> 5;
  const int wm   = w & 3;    // 32-row strip
  const int wn   = w >> 2;   // 64-col strip
  const size_t m0 = (size_t)blockIdx.y * 128;
  const size_t n0 = (size_t)blockIdx.x * 128;

  // Stage one 128x64 A tile + 128x64 Bt tile: 8 async b128 ops per thread.
  auto issue = [&](int buf, int k0) {
#pragma unroll
    for (int i = 0; i < 4; ++i) {
      int id  = tid + i * 256;        // 0..1023
      int row = id >> 3;              // 0..127
      int ch  = (id & 7) * 8;         // bf16 element offset, 16B chunks
      async_ld_b128(&As[buf][row * 64 + ch], A + (m0 + row) * 1024 + k0 + ch);
      async_ld_b128(&Bs[buf][row * 64 + ch], Bt + (n0 + row) * 1024 + k0 + ch);
    }
  };

  v8f acc[2][4];
  for (int i = 0; i < 2; ++i)
    for (int j = 0; j < 4; ++j)
      for (int e = 0; e < 8; ++e) acc[i][j][e] = 0.f;

  issue(0, 0);
  for (int step = 0; step < 16; ++step) {
    const int buf = step & 1;
    if (step + 1 < 16) {
      issue(buf ^ 1, (step + 1) * 64);
      wait_asynccnt<8>();          // this wave's stage `buf` copies complete
    } else {
      wait_asynccnt<0>();
    }
    __syncthreads();               // all waves' copies for `buf` visible

#pragma unroll
    for (int ks = 0; ks < 2; ++ks) {
      Frag a[2], b[4];
#pragma unroll
      for (int ms = 0; ms < 2; ++ms) {
        const __bf16* p = &As[buf][(wm * 32 + ms * 16 + ln) * 64 + ks * 32];
        a[ms].q[0] = *(const u32x4*)(p + g * 8);
        a[ms].q[1] = *(const u32x4*)(p + 16 + g * 8);
      }
#pragma unroll
      for (int ns = 0; ns < 4; ++ns) {
        const __bf16* p = &Bs[buf][(wn * 64 + ns * 16 + ln) * 64 + ks * 32];
        b[ns].q[0] = *(const u32x4*)(p + g * 8);
        b[ns].q[1] = *(const u32x4*)(p + 16 + g * 8);
      }
#pragma unroll
      for (int ms = 0; ms < 2; ++ms)
#pragma unroll
        for (int ns = 0; ns < 4; ++ns)
          acc[ms][ns] = wmma_bf16(a[ms].v, b[ns].v, acc[ms][ns]);
    }
    __syncthreads();               // buffer free before it is re-filled
  }

#pragma unroll
  for (int ns = 0; ns < 4; ++ns) {
    int col = (int)n0 + wn * 64 + ns * 16 + ln;
    float bv = bias[col];
#pragma unroll
    for (int ms = 0; ms < 2; ++ms) {
#pragma unroll
      for (int r = 0; r < 8; ++r) {
        int row = (int)m0 + wm * 32 + ms * 16 + r + 8 * g;
        Y[(size_t)row * 1024 + col] = (OutT)(acc[ms][ns][r] + bv);
      }
    }
  }
}

// ---------------------------------------------------------------------------
// V transpose: v[(b*2048+t)*1024 + h*64+d] -> vt[((b*16+h)*64+d)*2048 + t]
// ---------------------------------------------------------------------------
__global__ __launch_bounds__(256)
void transpose_v(const __bf16* __restrict__ vsrc, __bf16* __restrict__ vt) {
  size_t idx = (size_t)blockIdx.x * 256 + threadIdx.x;   // 4M elements
  int    t   = (int)(idx & 2047);
  size_t r   = idx >> 11;
  int    d   = (int)(r & 63);
  size_t bh  = r >> 6;
  int    h   = (int)(bh & 15);
  size_t b   = bh >> 4;
  vt[idx] = vsrc[(b * 2048 + t) * 1024 + h * 64 + d];
}

// ---------------------------------------------------------------------------
// Bhat GEMM (per b,h): Bhat[s][j] = q[b,h,s,:] . Qr[h,j,:]  (K = 64), f32 out.
// ---------------------------------------------------------------------------
__global__ __launch_bounds__(256)
void bhat_gemm(const __bf16* __restrict__ qb, const __bf16* __restrict__ Qr,
               float* __restrict__ Bhat) {
  const int lane = threadIdx.x & 31;
  const int ln = lane & 15, g = lane >> 4;
  int wid = blockIdx.x * 8 + (threadIdx.x >> 5);   // 65536 waves total
  int jt  = wid & 31;
  int st  = (wid >> 5) & 63;
  int bh  = wid >> 11;
  int b = bh >> 4, h = bh & 15;

  Frag a[2];
  {
    const __bf16* p = qb + (size_t)(b * 1024 + st * 16 + ln) * 1024 + h * 64;
#pragma unroll
    for (int ds = 0; ds < 2; ++ds) {
      a[ds].q[0] = *(const u32x4*)(p + ds * 32 + g * 8);
      a[ds].q[1] = *(const u32x4*)(p + ds * 32 + 16 + g * 8);
    }
  }
#pragma unroll
  for (int jn = 0; jn < 4; ++jn) {
    const __bf16* p = Qr + (size_t)(jt * 64 + jn * 16 + ln) * 1024 + h * 64;
    Frag b0, b1;
    b0.q[0] = *(const u32x4*)(p + g * 8);
    b0.q[1] = *(const u32x4*)(p + 16 + g * 8);
    b1.q[0] = *(const u32x4*)(p + 32 + g * 8);
    b1.q[1] = *(const u32x4*)(p + 32 + 16 + g * 8);
    v8f acc;
    for (int e = 0; e < 8; ++e) acc[e] = 0.f;
    acc = wmma_bf16(a[0].v, b0.v, acc);
    acc = wmma_bf16(a[1].v, b1.v, acc);
    int col = jt * 64 + jn * 16 + ln;
#pragma unroll
    for (int r = 0; r < 8; ++r) {
      int srow = st * 16 + r + 8 * g;
      Bhat[((size_t)bh * 1024 + srow) * 2048 + col] = acc[r];
    }
  }
}

// ---------------------------------------------------------------------------
// Fused attention (per wave = one b,h,16-query tile): S^T = k.q^T via WMMA,
// exact Transformer-XL shift gathered from Bhat, online softmax, PV via WMMA
// (S^T D-fragments are bit-identical to the PV A-fragment layout).
// ---------------------------------------------------------------------------
__global__ __launch_bounds__(256)
void attn_flash(const __bf16* __restrict__ qb, const __bf16* __restrict__ kb,
                const __bf16* __restrict__ vt, const float* __restrict__ Bhat,
                const float* __restrict__ mask, __bf16* __restrict__ attn) {
  const int lane = threadIdx.x & 31;
  const int ln = lane & 15, g = lane >> 4;
  int wid = blockIdx.x * 8 + (threadIdx.x >> 5);   // 2048 waves
  int st  = wid & 63;
  int bh  = wid >> 6;
  int b = bh >> 4, h = bh & 15;
  const int s0 = st * 16;
  const int s  = s0 + ln;

  Frag qf[2];
  {
    const __bf16* p = qb + (size_t)(b * 1024 + s) * 1024 + h * 64;
#pragma unroll
    for (int ds = 0; ds < 2; ++ds) {
      qf[ds].q[0] = *(const u32x4*)(p + ds * 32 + g * 8);
      qf[ds].q[1] = *(const u32x4*)(p + ds * 32 + 16 + g * 8);
    }
  }

  float m = -3.0e38f, l = 0.f;
  v8f acc[4];
  for (int f = 0; f < 4; ++f)
    for (int e = 0; e < 8; ++e) acc[f][e] = 0.f;

  const float* bh_base = Bhat + (size_t)bh * 1024 * 2048;

  for (int t0 = 0; t0 < 2048; t0 += 32) {
    Frag ka[2][2];
#pragma unroll
    for (int tt = 0; tt < 2; ++tt) {
      const __bf16* p =
          kb + (size_t)(b * 2048 + t0 + tt * 16 + ln) * 1024 + h * 64;
#pragma unroll
      for (int ds = 0; ds < 2; ++ds) {
        ka[tt][ds].q[0] = *(const u32x4*)(p + ds * 32 + g * 8);
        ka[tt][ds].q[1] = *(const u32x4*)(p + ds * 32 + 16 + g * 8);
      }
    }
    v8f stt[2];
#pragma unroll
    for (int tt = 0; tt < 2; ++tt) {
      v8f z;
      for (int e = 0; e < 8; ++e) z[e] = 0.f;
      z = wmma_bf16(ka[tt][0].v, qf[0].v, z);
      z = wmma_bf16(ka[tt][1].v, qf[1].v, z);
      stt[tt] = z;
    }

    float sc[16];
#pragma unroll
    for (int tt = 0; tt < 2; ++tt) {
#pragma unroll
      for (int r = 0; r < 8; ++r) {
        int t  = t0 + tt * 16 + r + 8 * g;
        int dt = t - s;
        float bval;
        if (dt == 1025)      bval = 0.f;
        else if (dt < 1025)  bval = bh_base[(size_t)s * 2048 + (dt + 1023)];
        else                 bval = bh_base[(size_t)(s + 1) * 2048 + (dt - 1026)];
        float mval = mask[(size_t)s * 2048 + t];
        sc[tt * 8 + r] = (stt[tt][r] + bval) * 0.125f + mval * -1.0e9f;
      }
    }

    float mloc = sc[0];
#pragma unroll
    for (int i = 1; i < 16; ++i) mloc = fmaxf(mloc, sc[i]);
    mloc = fmaxf(mloc, __shfl_xor(mloc, 16, 32));
    float mnew = fmaxf(m, mloc);
    float p[16], sloc = 0.f;
#pragma unroll
    for (int i = 0; i < 16; ++i) { p[i] = __expf(sc[i] - mnew); sloc += p[i]; }
    float ssum  = sloc + __shfl_xor(sloc, 16, 32);
    float scale = __expf(m - mnew);
    l = l * scale + ssum;
    m = mnew;

#pragma unroll
    for (int r = 0; r < 8; ++r) {
      float scr = __shfl(scale, r + 8 * g, 32);
#pragma unroll
      for (int f = 0; f < 4; ++f) acc[f][r] *= scr;
    }

    Frag pa;
#pragma unroll
    for (int i = 0; i < 16; ++i) pa.e[i] = (__bf16)p[i];

#pragma unroll
    for (int f = 0; f < 4; ++f) {
      const __bf16* vp =
          vt + (size_t)(bh * 64 + f * 16 + ln) * 2048 + t0 + g * 8;
      Frag vb;
      vb.q[0] = *(const u32x4*)(vp);
      vb.q[1] = *(const u32x4*)(vp + 16);
      acc[f] = wmma_bf16(pa.v, vb.v, acc[f]);
    }
  }

  float linv = 1.0f / l;
#pragma unroll
  for (int r = 0; r < 8; ++r) {
    float li  = __shfl(linv, r + 8 * g, 32);
    int  srow = s0 + r + 8 * g;
#pragma unroll
    for (int f = 0; f < 4; ++f)
      attn[(size_t)(b * 1024 + srow) * 1024 + h * 64 + f * 16 + ln] =
          (__bf16)(acc[f][r] * li);
  }
}

// ---------------------------------------------------------------------------
// Host-side orchestration
// ---------------------------------------------------------------------------
extern "C" void kernel_launch(void* const* d_in, const int* in_sizes, int n_in,
                              void* d_out, int out_size, void* d_ws, size_t ws_size,
                              hipStream_t stream) {
  const float* x    = (const float*)d_in[0];
  const float* memp = (const float*)d_in[1];
  const float* mask = (const float*)d_in[2];
  const float* rel  = (const float*)d_in[3];
  const float* Wq   = (const float*)d_in[4];
  const float* bq   = (const float*)d_in[5];
  const float* Wke  = (const float*)d_in[6];
  const float* bke  = (const float*)d_in[7];
  const float* Wkr  = (const float*)d_in[8];
  const float* bkr  = (const float*)d_in[9];
  const float* Wv   = (const float*)d_in[10];
  const float* bv   = (const float*)d_in[11];
  const float* Wo   = (const float*)d_in[12];
  const float* bo   = (const float*)d_in[13];
  float* out = (float*)d_out;

  // Workspace layout (MB offsets), total ~318 MB
  char*   ws    = (char*)d_ws;
  __bf16* xb    = (__bf16*)(ws);                            //  4 MB  x bf16
  __bf16* xtb   = (__bf16*)(ws + ((size_t)4  << 20));       //  8 MB  concat bf16
  __bf16* relb  = (__bf16*)(ws + ((size_t)12 << 20));       //  4 MB
  __bf16* Wqt   = (__bf16*)(ws + ((size_t)16 << 20));       //  2 MB each
  __bf16* Wket  = (__bf16*)(ws + ((size_t)18 << 20));
  __bf16* Wkrt  = (__bf16*)(ws + ((size_t)20 << 20));
  __bf16* Wvt   = (__bf16*)(ws + ((size_t)22 << 20));
  __bf16* Wot   = (__bf16*)(ws + ((size_t)24 << 20));
  __bf16* qb    = (__bf16*)(ws + ((size_t)26 << 20));       //  4 MB
  __bf16* kb    = (__bf16*)(ws + ((size_t)30 << 20));       //  8 MB
  __bf16* vb_   = (__bf16*)(ws + ((size_t)38 << 20));       //  8 MB
  __bf16* Qr    = (__bf16*)(ws + ((size_t)46 << 20));       //  4 MB
  __bf16* vtb   = (__bf16*)(ws + ((size_t)50 << 20));       //  8 MB
  __bf16* attnb = (__bf16*)(ws + ((size_t)58 << 20));       //  4 MB
  float*  Bhat  = (float*) (ws + ((size_t)62 << 20));       // 256 MB

  // bf16 conversions (concat fused; weights transposed for async DMA staging)
  convert_bf16<<<8192, 256, 0, stream>>>(x, xb);       // 2M elems
  convert_bf16<<<8192, 256, 0, stream>>>(rel, relb);   // 2M elems
  build_xtb<<<16384, 256, 0, stream>>>(x, memp, xtb);  // 4M elems
  wtrans_bf16<<<dim3(32, 32), 256, 0, stream>>>(Wq,  Wqt);
  wtrans_bf16<<<dim3(32, 32), 256, 0, stream>>>(Wke, Wket);
  wtrans_bf16<<<dim3(32, 32), 256, 0, stream>>>(Wkr, Wkrt);
  wtrans_bf16<<<dim3(32, 32), 256, 0, stream>>>(Wv,  Wvt);
  wtrans_bf16<<<dim3(32, 32), 256, 0, stream>>>(Wo,  Wot);

  // Projections (async-DMA staged bf16 WMMA GEMMs)
  gemm1024_bf16<__bf16><<<dim3(8, 16), 256, 0, stream>>>(xb,  Wqt,  bq,  qb);
  gemm1024_bf16<__bf16><<<dim3(8, 32), 256, 0, stream>>>(xtb, Wket, bke, kb);
  gemm1024_bf16<__bf16><<<dim3(8, 32), 256, 0, stream>>>(xtb, Wvt,  bv,  vb_);
  gemm1024_bf16<__bf16><<<dim3(8, 16), 256, 0, stream>>>(relb, Wkrt, bkr, Qr);

  transpose_v<<<16384, 256, 0, stream>>>(vb_, vtb);

  // Relative scores (unshifted), f32
  bhat_gemm<<<8192, 256, 0, stream>>>(qb, Qr, Bhat);

  // Fused scores + exact rel-shift + softmax + PV
  attn_flash<<<256, 256, 0, stream>>>(qb, kb, vtb, Bhat, mask, attnb);

  // Output projection (f32 out)
  gemm1024_bf16<float><<<dim3(8, 16), 256, 0, stream>>>(attnb, Wot, bo, out);
}
